// ROIHead_21749714387037
// MI455X (gfx1250) — matrix operand order repaired and checked by reading, not compiled, not run on hardware
//
#include <hip/hip_runtime.h>
#include <hip/hip_bf16.h>
#include <math.h>

// ---------------- problem constants (from reference setup) ----------------
#define P_ROIS 300
#define C_FEAT 512
#define FH 50
#define FW 67
#define POOL 7
#define FIN (C_FEAT * POOL * POOL)   // 25088
#define DHID 1024
#define NUM_CLASSES 21
#define NHEAD 112                    // 21 cls + 84 reg, padded to 112 (7 tiles)
#define IMG_W 1072.0f
#define IMG_H 800.0f
#define SCALE (1.0f / 16.0f)
#define NMS_THR 0.5f
#define SCORE_THR 0.05f
#define MIN_SIZE 1.0f
#define TOPK 100
#define BBOX_CLIP 4.135166556742356f // log(1000/16)
#define MPAD 320                     // 300 rounded to 32 (2-M-tile blocks divide evenly)
#define KSPLIT 8                     // FC6 K-split: 25088 / 8 = 3136 = 98 k-steps
#define NBOX 6000                    // 300 * 20
#define SORT_N 8192

typedef __attribute__((ext_vector_type(8)))  __bf16 v8bf;
typedef __attribute__((ext_vector_type(16))) __bf16 v16bf;
typedef __attribute__((ext_vector_type(8)))  float  v8f;

// ---------------- small utility kernels ----------------
__global__ void k_f2bf(const float* __restrict__ src, __bf16* __restrict__ dst, int n) {
  int i = blockIdx.x * blockDim.x + threadIdx.x;
  if (i < n) dst[i] = (__bf16)src[i];
}

__global__ void k_zero_bf(__bf16* __restrict__ dst, int n) {
  int i = blockIdx.x * blockDim.x + threadIdx.x;
  if (i < n) dst[i] = (__bf16)0.0f;
}

// build fused head weight [112][1024] (rows 0..20 cls, 21..104 reg, rest 0) + bias[112]
__global__ void k_build_heads(const float* __restrict__ cls_w, const float* __restrict__ reg_w,
                              const float* __restrict__ cls_b, const float* __restrict__ reg_b,
                              __bf16* __restrict__ WH, float* __restrict__ bH) {
  int i = blockIdx.x * blockDim.x + threadIdx.x;
  if (i < NHEAD * DHID) {
    int n = i >> 10, k = i & 1023;
    float v = 0.0f;
    if (n < 21)        v = cls_w[n * DHID + k];
    else if (n < 105)  v = reg_w[(n - 21) * DHID + k];
    WH[i] = (__bf16)v;
  }
  if (i < NHEAD) {
    float b = 0.0f;
    if (i < 21)       b = cls_b[i];
    else if (i < 105) b = reg_b[i - 21];
    bH[i] = b;
  }
}

// ---------------- ROI-Align (POOL=7, sampling_ratio=2) ----------------
// Writes pooled activations directly as bf16 GEMM-A matrix: A[p][c*49 + py*7 + px]
__global__ void k_roi_align(const float* __restrict__ feat, const float* __restrict__ rois,
                            __bf16* __restrict__ A) {
  int p = blockIdx.x;
  int e = blockIdx.y * blockDim.x + threadIdx.x;
  if (e >= FIN) return;
  int c  = e / 49;
  int r  = e - c * 49;
  int py = r / POOL;
  int px = r - py * POOL;

  float rx0 = rois[p * 4 + 0], ry0 = rois[p * 4 + 1];
  float rx1 = rois[p * 4 + 2], ry1 = rois[p * 4 + 3];
  float x0r = rx0 * SCALE - 0.5f;
  float y0r = ry0 * SCALE - 0.5f;
  float bw = (rx1 - rx0) * SCALE / (float)POOL;
  float bh = (ry1 - ry0) * SCALE / (float)POOL;

  const float* f = feat + (size_t)c * (FH * FW);
  float acc = 0.0f;
#pragma unroll
  for (int iy = 0; iy < 2; ++iy) {
#pragma unroll
    for (int ix = 0; ix < 2; ++ix) {
      float ty = ((float)(py * 2 + iy) + 0.5f) * 0.5f;
      float tx = ((float)(px * 2 + ix) + 0.5f) * 0.5f;
      float y = y0r + bh * ty;
      float x = x0r + bw * tx;
      bool valid = (y >= -1.0f) && (y <= (float)FH) && (x >= -1.0f) && (x <= (float)FW);
      float yc = fminf(fmaxf(y, 0.0f), (float)(FH - 1));
      float xc = fminf(fmaxf(x, 0.0f), (float)(FW - 1));
      int yl = (int)floorf(yc), xl = (int)floorf(xc);
      int yh = min(yl + 1, FH - 1), xh = min(xl + 1, FW - 1);
      float ly = yc - (float)yl, lx = xc - (float)xl;
      float hy = 1.0f - ly, hx = 1.0f - lx;
      float v = f[yl * FW + xl] * hy * hx + f[yl * FW + xh] * hy * lx +
                f[yh * FW + xl] * ly * hx + f[yh * FW + xh] * ly * lx;
      acc += valid ? v : 0.0f;
    }
  }
  acc *= 0.25f;
  A[(size_t)p * FIN + e] = (__bf16)acc;
}

// ---------------- fragment load helpers (contiguous k order, same for A and B) ----------------
// A fragment (16-bit 16x32): lane l (l&15 = M-row), half h=l>>4 holds
//   K = 16h..16h+15 contiguous (matches B fragment) -> one 32B load.
__device__ __forceinline__ v16bf load_afrag(const __bf16* arow, int k, int h) {
  return *(const v16bf*)(arow + k + 16 * h);
}
// B fragment (32x16, K x N): lane l holds column n=l&15, K = 16h..16h+15 contiguous
//   -> one 32B load straight from row-major W[N][K].
__device__ __forceinline__ v16bf load_bfrag(const __bf16* brow, int k, int h) {
  return *(const v16bf*)(brow + k + 16 * h);
}

// ---------------- blocked WMMA GEMM for FC6 (2M x 2N tiles, K-split) ---------
// One wave computes a 32x32 output block: 2 A frags x 2 B frags -> 4 WMMAs per
// k-step, each fragment reused twice (halves L2 traffic vs 16x16/wave).
// grid = (N/32, MPAD/32, KSPLIT); partials -> part[kc][MPAD][N] (fp32),
// reduced deterministically by k_reduce.
__global__ __launch_bounds__(32)
void k_gemm_bf16_blk(const __bf16* __restrict__ A, int lda,
                     const __bf16* __restrict__ W, int ldw,
                     float* __restrict__ part, int N, int K) {
  int lane = threadIdx.x & 31;
  int h   = lane >> 4;
  int l15 = lane & 15;
  int nt0 = blockIdx.x * 2, mt0 = blockIdx.y * 2, kc = blockIdx.z;
  int kchunk = K / KSPLIT;
  int kbeg = kc * kchunk, kend = kbeg + kchunk;

  const __bf16* arow0 = A + (size_t)(mt0 * 16 + l15) * lda;
  const __bf16* arow1 = arow0 + (size_t)16 * lda;
  const __bf16* brow0 = W + (size_t)(nt0 * 16 + l15) * ldw;
  const __bf16* brow1 = brow0 + (size_t)16 * ldw;

  v8f acc00 = {}, acc01 = {}, acc10 = {}, acc11 = {};
  for (int k = kbeg; k < kend; k += 32) {
    __builtin_prefetch(brow0 + k + 512, 0, 1);
    __builtin_prefetch(brow1 + k + 512, 0, 1);
    v16bf a0 = load_afrag(arow0, k, h);
    v16bf a1 = load_afrag(arow1, k, h);
    v16bf b0 = load_bfrag(brow0, k, h);
    v16bf b1 = load_bfrag(brow1, k, h);
    acc00 = __builtin_amdgcn_wmma_f32_16x16x32_bf16(false, a0, false, b0, (short)0, acc00, false, false);
    acc01 = __builtin_amdgcn_wmma_f32_16x16x32_bf16(false, a0, false, b1, (short)0, acc01, false, false);
    acc10 = __builtin_amdgcn_wmma_f32_16x16x32_bf16(false, a1, false, b0, (short)0, acc10, false, false);
    acc11 = __builtin_amdgcn_wmma_f32_16x16x32_bf16(false, a1, false, b1, (short)0, acc11, false, false);
  }

  // C/D layout: VGPR r -> M = r + 8h, N = lane&15
  float* pk = part + (size_t)kc * MPAD * N;
#pragma unroll
  for (int r = 0; r < 8; ++r) {
    int m0 = mt0 * 16 + r + 8 * h;
    int n0 = nt0 * 16 + l15;
    pk[(size_t)m0 * N + n0]               = acc00[r];
    pk[(size_t)m0 * N + n0 + 16]          = acc01[r];
    pk[(size_t)(m0 + 16) * N + n0]        = acc10[r];
    pk[(size_t)(m0 + 16) * N + n0 + 16]   = acc11[r];
  }
}

// deterministic K-split reduction + bias + relu -> fp32 and bf16 activations
__global__ void k_reduce(const float* __restrict__ part, const float* __restrict__ bias,
                         float* __restrict__ outF, __bf16* __restrict__ outB, int N, int M) {
  int i = blockIdx.x * blockDim.x + threadIdx.x;
  if (i >= MPAD * N) return;
  int m = i / N, n = i - m * N;
  float v = 0.0f;
#pragma unroll
  for (int kc = 0; kc < KSPLIT; ++kc) v += part[(size_t)kc * MPAD * N + i];
  v = fmaxf(v + bias[n], 0.0f);
  if (m >= M) v = 0.0f;  // keep pad rows clean for the next GEMM
  outF[i] = v;
  outB[i] = (__bf16)v;
}

// ---------------- simple WMMA GEMM (one 16x16 tile per wave) ----------------
// out[M][N] = act(A[M][K] @ W[N][K]^T + bias). Used for FC7 and the heads.
__global__ __launch_bounds__(32)
void k_gemm_bf16(const __bf16* __restrict__ A, int lda,
                 const __bf16* __restrict__ W, int ldw,
                 const float* __restrict__ bias,
                 float* __restrict__ outF, __bf16* __restrict__ outB, int ldo,
                 int M, int K, int relu) {
  int lane = threadIdx.x & 31;
  int h   = lane >> 4;
  int l15 = lane & 15;
  int nt = blockIdx.x, mt = blockIdx.y;

  const __bf16* arow = A + (size_t)(mt * 16 + l15) * lda;
  const __bf16* brow = W + (size_t)(nt * 16 + l15) * ldw;

  v8f acc = {};
  for (int k = 0; k < K; k += 32) {
    __builtin_prefetch(brow + k + 512, 0, 1);
    __builtin_prefetch(arow + k + 512, 0, 1);
    v16bf a = load_afrag(arow, k, h);
    v16bf b = load_bfrag(brow, k, h);
    acc = __builtin_amdgcn_wmma_f32_16x16x32_bf16(false, a, false, b, (short)0, acc, false, false);
  }

  int n = nt * 16 + l15;
  float bn = bias[n];
#pragma unroll
  for (int r = 0; r < 8; ++r) {
    int m = mt * 16 + r + 8 * h;
    float v = acc[r] + bn;
    if (relu) v = fmaxf(v, 0.0f);
    if (m >= M) v = 0.0f;  // zero-pad rows so downstream GEMMs see clean A
    outF[(size_t)m * ldo + n] = v;
    if (outB) outB[(size_t)m * ldo + n] = (__bf16)v;
  }
}

// ---------------- decode boxes + softmax, produce flat 6000-entry lists ------
__global__ void k_decode(const float* __restrict__ H /*[MPAD][112]*/,
                         const float* __restrict__ proposals,
                         float* __restrict__ boxes6k, float* __restrict__ scores6k,
                         int* __restrict__ labels6k, float* __restrict__ skey6k) {
  int p = blockIdx.x * blockDim.x + threadIdx.x;
  if (p >= P_ROIS) return;
  const float* hrow = H + (size_t)p * NHEAD;

  // softmax over 21 logits
  float mx = hrow[0];
  for (int c = 1; c < NUM_CLASSES; ++c) mx = fmaxf(mx, hrow[c]);
  float ex[NUM_CLASSES];
  float sum = 0.0f;
  for (int c = 0; c < NUM_CLASSES; ++c) { ex[c] = expf(hrow[c] - mx); sum += ex[c]; }
  float inv = 1.0f / sum;

  float px0 = proposals[p * 4 + 0], py0 = proposals[p * 4 + 1];
  float px1 = proposals[p * 4 + 2], py1 = proposals[p * 4 + 3];
  float w = px1 - px0, hh = py1 - py0;
  float cx = px0 + 0.5f * w, cy = py0 + 0.5f * hh;

  for (int c = 1; c < NUM_CLASSES; ++c) {
    const float* pr = hrow + NUM_CLASSES + c * 4;
    float dx = pr[0], dy = pr[1];
    float dw = fminf(pr[2], BBOX_CLIP);
    float dh = fminf(pr[3], BBOX_CLIP);
    float pcx = dx * w + cx, pcy = dy * hh + cy;
    float pw = expf(dw) * w, ph = expf(dh) * hh;
    float bx0 = fminf(fmaxf(pcx - 0.5f * pw, 0.0f), IMG_W);
    float by0 = fminf(fmaxf(pcy - 0.5f * ph, 0.0f), IMG_H);
    float bx1 = fminf(fmaxf(pcx + 0.5f * pw, 0.0f), IMG_W);
    float by1 = fminf(fmaxf(pcy + 0.5f * ph, 0.0f), IMG_H);
    int idx = p * (NUM_CLASSES - 1) + (c - 1);
    boxes6k[idx * 4 + 0] = bx0;
    boxes6k[idx * 4 + 1] = by0;
    boxes6k[idx * 4 + 2] = bx1;
    boxes6k[idx * 4 + 3] = by1;
    float sc = ex[c] * inv;
    scores6k[idx] = sc;
    labels6k[idx] = c - 1;
    bool valid = (sc > SCORE_THR) && ((bx1 - bx0) >= MIN_SIZE) && ((by1 - by0) >= MIN_SIZE);
    skey6k[idx] = valid ? sc : -__builtin_inff();
  }
}

// ---------------- bitonic sort (descending key, ascending index tiebreak) ----
__global__ __launch_bounds__(1024)
void k_sort(const float* __restrict__ skey, float* __restrict__ okey, int* __restrict__ oidx) {
  __shared__ float kk[SORT_N];
  __shared__ int   id[SORT_N];
  for (int i = threadIdx.x; i < SORT_N; i += 1024) {
    kk[i] = (i < NBOX) ? skey[i] : -__builtin_inff();
    id[i] = i;
  }
  __syncthreads();
  for (int size = 2; size <= SORT_N; size <<= 1) {
    for (int stride = size >> 1; stride > 0; stride >>= 1) {
      for (int i = threadIdx.x; i < SORT_N; i += 1024) {
        int j = i ^ stride;
        if (j > i) {
          bool descBlock = ((i & size) == 0);
          float ki = kk[i], kj = kk[j];
          int ii = id[i], ij = id[j];
          bool iFirst = (ki > kj) || (ki == kj && ii < ij); // stable descending
          bool doSwap = descBlock ? !iFirst : iFirst;
          if (doSwap) { kk[i] = kj; kk[j] = ki; id[i] = ij; id[j] = ii; }
        }
      }
      __syncthreads();
    }
  }
  for (int i = threadIdx.x; i < SORT_N; i += 1024) { okey[i] = kk[i]; oidx[i] = id[i]; }
}

// ---------------- NMS + final top-100 selection --------------------------
__global__ __launch_bounds__(1024)
void k_nms_final(const float* __restrict__ boxes6k, const int* __restrict__ labels6k,
                 const float* __restrict__ skey_sorted, const int* __restrict__ idx_sorted,
                 float* __restrict__ ob /*[NBOX][4] offset boxes*/, float* __restrict__ area,
                 float* __restrict__ out) {
  __shared__ int keep[NBOX];
  int tid = threadIdx.x;

  // precompute class-offset boxes + areas + initial keep (=valid) in sorted order
  for (int i = tid; i < NBOX; i += 1024) {
    int oi = idx_sorted[i];
    float x1 = 0, y1 = 0, x2 = 0, y2 = 0;
    int kp = 0;
    if (oi < NBOX) {
      float off = 2048.0f * (float)labels6k[oi];  // > IMG_W so classes never overlap
      x1 = boxes6k[oi * 4 + 0] + off;
      y1 = boxes6k[oi * 4 + 1] + off;
      x2 = boxes6k[oi * 4 + 2] + off;
      y2 = boxes6k[oi * 4 + 3] + off;
      kp = (skey_sorted[i] > -1e37f) ? 1 : 0;
    }
    ob[i * 4 + 0] = x1; ob[i * 4 + 1] = y1; ob[i * 4 + 2] = x2; ob[i * 4 + 3] = y2;
    area[i] = (x2 - x1) * (y2 - y1);
    keep[i] = kp;
  }
  __syncthreads();

  // greedy NMS: sequential over i, parallel suppression over j > i
  for (int i = 0; i < NBOX; ++i) {
    if (keep[i]) {
      float ax1 = ob[i * 4 + 0], ay1 = ob[i * 4 + 1];
      float ax2 = ob[i * 4 + 2], ay2 = ob[i * 4 + 3];
      float aa = area[i];
      for (int j = i + 1 + tid; j < NBOX; j += 1024) {
        if (keep[j]) {
          float iw = fmaxf(fminf(ax2, ob[j * 4 + 2]) - fmaxf(ax1, ob[j * 4 + 0]), 0.0f);
          float ih = fmaxf(fminf(ay2, ob[j * 4 + 3]) - fmaxf(ay1, ob[j * 4 + 1]), 0.0f);
          float inter = iw * ih;
          float iou = inter / (aa + area[j] - inter);
          if (iou > NMS_THR) keep[j] = 0;
        }
      }
    }
    __syncthreads();
  }

  // final selection: first TOPK kept entries in sorted order
  if (tid == 0) {
    int* outLab = (int*)out;  // labels region written as int32 bit pattern
    int count = 0;
    for (int i = 0; i < NBOX && count < TOPK; ++i) {
      if (keep[i]) {
        int oi = idx_sorted[i];
        out[count * 4 + 0] = boxes6k[oi * 4 + 0];
        out[count * 4 + 1] = boxes6k[oi * 4 + 1];
        out[count * 4 + 2] = boxes6k[oi * 4 + 2];
        out[count * 4 + 3] = boxes6k[oi * 4 + 3];
        outLab[4 * TOPK + count] = labels6k[oi];
        out[5 * TOPK + count] = skey_sorted[i];
        ++count;
      }
    }
    for (; count < TOPK; ++count) {
      out[count * 4 + 0] = 0.0f; out[count * 4 + 1] = 0.0f;
      out[count * 4 + 2] = 0.0f; out[count * 4 + 3] = 0.0f;
      outLab[4 * TOPK + count] = -1;
      out[5 * TOPK + count] = 0.0f;
    }
  }
}

// ---------------- host-side launch --------------------------------------
extern "C" void kernel_launch(void* const* d_in, const int* in_sizes, int n_in,
                              void* d_out, int out_size, void* d_ws, size_t ws_size,
                              hipStream_t stream) {
  (void)in_sizes; (void)n_in; (void)out_size; (void)ws_size;
  const float* feat      = (const float*)d_in[1];   // (1, 512, 50, 67)
  const float* proposals = (const float*)d_in[2];   // (300, 4)
  const float* fc6_w     = (const float*)d_in[3];   // (1024, 25088)
  const float* fc6_b     = (const float*)d_in[4];
  const float* fc7_w     = (const float*)d_in[5];   // (1024, 1024)
  const float* fc7_b     = (const float*)d_in[6];
  const float* cls_w     = (const float*)d_in[7];   // (21, 1024)
  const float* cls_b     = (const float*)d_in[8];
  const float* reg_w     = (const float*)d_in[9];   // (84, 1024)
  const float* reg_b     = (const float*)d_in[10];
  float* out = (float*)d_out;

  // workspace carve-out (256B aligned)
  uintptr_t base = (uintptr_t)d_ws;
  size_t off = 0;
  auto carve = [&](size_t bytes) -> uintptr_t {
    uintptr_t p = base + off;
    off = (off + bytes + 255) & ~(size_t)255;
    return p;
  };
  __bf16* A_bf   = (__bf16*)carve((size_t)MPAD * FIN * 2);          // pooled activations
  __bf16* W6_bf  = (__bf16*)carve((size_t)DHID * FIN * 2);
  float*  part   = (float*)carve((size_t)KSPLIT * MPAD * DHID * 4); // FC6 K-split partials
  float*  X1     = (float*)carve((size_t)MPAD * DHID * 4);
  __bf16* X1_bf  = (__bf16*)carve((size_t)MPAD * DHID * 2);
  __bf16* W7_bf  = (__bf16*)carve((size_t)DHID * DHID * 2);
  float*  X2     = (float*)carve((size_t)MPAD * DHID * 4);
  __bf16* X2_bf  = (__bf16*)carve((size_t)MPAD * DHID * 2);
  __bf16* WH_bf  = (__bf16*)carve((size_t)NHEAD * DHID * 2);
  float*  bH     = (float*)carve((size_t)NHEAD * 4);
  float*  Hout   = (float*)carve((size_t)MPAD * NHEAD * 4);
  float*  boxes6 = (float*)carve((size_t)NBOX * 4 * 4);
  float*  score6 = (float*)carve((size_t)NBOX * 4);
  int*    label6 = (int*)carve((size_t)NBOX * 4);
  float*  skey6  = (float*)carve((size_t)NBOX * 4);
  float*  skk    = (float*)carve((size_t)SORT_N * 4);
  int*    sid    = (int*)carve((size_t)SORT_N * 4);
  float*  obbuf  = (float*)carve((size_t)NBOX * 4 * 4);
  float*  area   = (float*)carve((size_t)NBOX * 4);

  // 1) weight conversions to bf16 (W6_bf = 51MB, L2-resident for GEMM re-reads)
  {
    int n = DHID * FIN;
    k_f2bf<<<(n + 255) / 256, 256, 0, stream>>>(fc6_w, W6_bf, n);
  }
  {
    int n = DHID * DHID;
    k_f2bf<<<(n + 255) / 256, 256, 0, stream>>>(fc7_w, W7_bf, n);
  }
  {
    int n = NHEAD * DHID;
    k_build_heads<<<(n + 255) / 256, 256, 0, stream>>>(cls_w, reg_w, cls_b, reg_b, WH_bf, bH);
  }
  // 2) zero pad rows 300..319 of the GEMM-A matrix
  {
    int n = (MPAD - P_ROIS) * FIN;
    k_zero_bf<<<(n + 255) / 256, 256, 0, stream>>>(A_bf + (size_t)P_ROIS * FIN, n);
  }
  // 3) ROI-Align -> bf16 A matrix
  {
    dim3 grid(P_ROIS, (FIN + 255) / 256);
    k_roi_align<<<grid, 256, 0, stream>>>(feat, proposals, A_bf);
  }
  // 4) FC6: [300,25088] @ [25088,1024], 2x2 register-blocked + K-split 8
  //    grid 32 x 10 x 8 = 2560 waves, 4 WMMAs per k-step per wave
  {
    dim3 grid(DHID / 32, MPAD / 32, KSPLIT);
    k_gemm_bf16_blk<<<grid, 32, 0, stream>>>(A_bf, FIN, W6_bf, FIN, part, DHID, FIN);
  }
  //    deterministic fixed-order reduction + bias + relu
  {
    int n = MPAD * DHID;
    k_reduce<<<(n + 255) / 256, 256, 0, stream>>>(part, fc6_b, X1, X1_bf, DHID, P_ROIS);
  }
  // 5) FC7: [300,1024] @ [1024,1024] + relu (simple 16x16/wave, 1280 waves)
  {
    dim3 grid(DHID / 16, MPAD / 16);
    k_gemm_bf16<<<grid, 32, 0, stream>>>(X1_bf, DHID, W7_bf, DHID, fc7_b,
                                         X2, X2_bf, DHID, P_ROIS, DHID, 1);
  }
  // 6) fused cls+reg heads: [300,1024] @ [1024,112]
  {
    dim3 grid(NHEAD / 16, MPAD / 16);
    k_gemm_bf16<<<grid, 32, 0, stream>>>(X2_bf, DHID, WH_bf, DHID, bH,
                                         Hout, (__bf16*)nullptr, NHEAD, P_ROIS, DHID, 0);
  }
  // 7) decode + softmax -> 6000 flat candidates
  k_decode<<<(P_ROIS + 127) / 128, 128, 0, stream>>>(Hout, proposals, boxes6, score6,
                                                     label6, skey6);
  // 8) stable descending sort of 6000 score keys (bitonic, single WG, 64KB LDS)
  k_sort<<<1, 1024, 0, stream>>>(skey6, skk, sid);
  // 9) greedy NMS + top-100 emit
  k_nms_final<<<1, 1024, 0, stream>>>(boxes6, label6, skk, sid, obbuf, area, out);
  (void)score6;
}
